// VectorQuantizer_63745904607523
// MI455X (gfx1250) — matrix operand order repaired and checked by reading, not compiled, not run on hardware
//
#include <hip/hip_runtime.h>
#include <math.h>

#define NUM_EMB   512
#define DIM       64
#define N_TOKENS  (32 * 4096)
#define TOK_PER_WG 256          // 8 waves x 32 tokens
#define COMMIT    0.25f

// d_out layout (floats): [quantized N*D][loss][perplexity][enc N][min_distance]
#define OFF_LOSS ((size_t)N_TOKENS * DIM)
#define OFF_PERP (OFF_LOSS + 1)
#define OFF_ENC  (OFF_LOSS + 2)
#define OFF_MIND (OFF_ENC + N_TOKENS)

typedef __attribute__((ext_vector_type(16))) __bf16 v16bf;
typedef __attribute__((ext_vector_type(4)))  __bf16 v4bf;
typedef __attribute__((ext_vector_type(8)))  float  v8f;

// ws layout (floats): hist[512], acc[0]=sum((q-z)^2), acc[1]=sum(min_dist)
__global__ void vq_zero_ws(float* __restrict__ ws) {
    int i = blockIdx.x * blockDim.x + threadIdx.x;
    if (i < NUM_EMB + 2) ws[i] = 0.0f;
}

// Build one 16x32 bf16 A fragment (hi/lo split) for this lane.
// Lane covers K = {kb..kb+15} of the 32-wide chunk (kb = half*16 + chunk*32).
__device__ __forceinline__ void make_afrag(const float* __restrict__ zrow, int kb,
                                           v16bf& hi, v16bf& lo, float& nrm) {
    const float4* p0 = (const float4*)(zrow + kb);
    const float4* p1 = (const float4*)(zrow + kb + 8);
    float4 a = p0[0], b = p0[1], c = p1[0], d = p1[1];
    float f[16] = {a.x, a.y, a.z, a.w, b.x, b.y, b.z, b.w,
                   c.x, c.y, c.z, c.w, d.x, d.y, d.z, d.w};
    #pragma unroll
    for (int i = 0; i < 16; ++i) {
        __bf16 h = (__bf16)f[i];
        hi[i] = h;
        lo[i] = (__bf16)(f[i] - (float)h);
        nrm  += f[i] * f[i];
    }
}

// Per-tile epilogue: cross-lane argmin, encodings/histogram/min-dist, gather+loss.
__device__ __forceinline__ void epilogue_tile(
    float* best, float* bidx, float zn, int tok0,
    int half, int l16,
    const float* __restrict__ z, const float* __restrict__ emb,
    float* __restrict__ out, float* __restrict__ hist, float* __restrict__ acc,
    float& ss)
{
    #pragma unroll
    for (int off = 8; off >= 1; off >>= 1) {
        #pragma unroll
        for (int v = 0; v < 8; ++v) {
            float ob = __shfl_xor(best[v], off, 32);
            float oi = __shfl_xor(bidx[v], off, 32);
            if (ob < best[v] || (ob == best[v] && oi < bidx[v])) {
                best[v] = ob; bidx[v] = oi;
            }
        }
    }
    float msum = 0.0f;
    #pragma unroll
    for (int v = 0; v < 8; ++v) {
        float zt = __shfl(zn, v + 8 * half, 32);   // ||z||^2 of this slot's token
        msum += best[v] + zt;
    }
    if (l16 == 0) {
        #pragma unroll
        for (int v = 0; v < 8; ++v) {
            int tok = tok0 + v + 8 * half;
            out[OFF_ENC + tok] = bidx[v];
            atomicAdd(&hist[(int)bidx[v]], 1.0f);
        }
        atomicAdd(&acc[1], msum);
    }
    #pragma unroll
    for (int v = 0; v < 8; ++v) {
        int tok  = tok0 + v + 8 * half;
        int code = (int)bidx[v];                       // uniform within half
        const float4* er  = (const float4*)(emb + code * DIM);  // exact f32 codebook
        const float4* zr4 = (const float4*)(z + (size_t)tok * DIM);
        float4 e4 = er[l16];
        float4 z4 = zr4[l16];
        float dx = e4.x - z4.x, dy = e4.y - z4.y;
        float dz = e4.z - z4.z, dw = e4.w - z4.w;
        ss += dx * dx + dy * dy + dz * dz + dw * dw;
        ((float4*)(out + (size_t)tok * DIM))[l16] = e4;
    }
}

__global__ __launch_bounds__(256, 1) void vq_main(
    const float* __restrict__ z, const float* __restrict__ emb,
    float* __restrict__ out, float* __restrict__ ws)
{
    extern __shared__ char smem_raw[];
    __bf16* s_eh    = (__bf16*)smem_raw;              // 512*64 bf16 (64 KB)
    __bf16* s_el    = s_eh + NUM_EMB * DIM;           // 512*64 bf16 (64 KB)
    float*  s_enorm = (float*)(s_el + NUM_EMB * DIM); // 512 f32

    float* hist = ws;
    float* acc  = ws + NUM_EMB;

    const int tid  = threadIdx.x;
    const int wave = tid >> 5;
    const int lane = tid & 31;
    const int half = lane >> 4;
    const int l16  = lane & 15;
    const int wave_tok0 = blockIdx.x * TOK_PER_WG + wave * 32;

    // ---- Split codebook into bf16 hi/lo in LDS ----
    {
        const float4* src = (const float4*)emb;
        for (int i = tid; i < NUM_EMB * DIM / 4; i += 256) {
            float4 v = src[i];
            float f[4] = {v.x, v.y, v.z, v.w};
            v4bf h, l;
            #pragma unroll
            for (int j = 0; j < 4; ++j) {
                __bf16 hb = (__bf16)f[j];
                h[j] = hb;
                l[j] = (__bf16)(f[j] - (float)hb);
            }
            *(v4bf*)(s_eh + i * 4) = h;
            *(v4bf*)(s_el + i * 4) = l;
        }
    }
    // ---- Exact f32 code norms ----
    for (int c = tid; c < NUM_EMB; c += 256) {
        const float4* e = (const float4*)(emb + c * DIM);
        float s = 0.0f;
        #pragma unroll
        for (int k = 0; k < DIM / 4; ++k) {
            float4 t = e[k];
            s += t.x * t.x + t.y * t.y + t.z * t.z + t.w * t.w;
        }
        s_enorm[c] = s;
    }
    __syncthreads();

    // ---- A fragments: 2 token tiles x 2 K-chunks x (hi,lo) ----
    v16bf ah[2][2], al[2][2];
    float zn0 = 0.0f, zn1 = 0.0f;
    const int kb = half * 16;
    {
        const float* zr0 = z + (size_t)(wave_tok0 + l16) * DIM;
        const float* zr1 = z + (size_t)(wave_tok0 + 16 + l16) * DIM;
        #pragma unroll
        for (int c = 0; c < 2; ++c) {
            make_afrag(zr0, c * 32 + kb, ah[0][c], al[0][c], zn0);
            make_afrag(zr1, c * 32 + kb, ah[1][c], al[1][c], zn1);
        }
        zn0 += __shfl_xor(zn0, 16, 32);
        zn1 += __shfl_xor(zn1, 16, 32);
    }

    float best0[8], bidx0[8], best1[8], bidx1[8];
    #pragma unroll
    for (int v = 0; v < 8; ++v) {
        best0[v] = 3.0e38f; bidx0[v] = 0.0f;
        best1[v] = 3.0e38f; bidx1[v] = 0.0f;
    }

    // ---- GEMM (split-bf16, 3-term) + fused argmin; B shared by both token tiles ----
    const int koff = half * 16;
    for (int ct = 0; ct < NUM_EMB / 16; ++ct) {
        const int code = ct * 16 + l16;
        v8f c0 = {0.f, 0.f, 0.f, 0.f, 0.f, 0.f, 0.f, 0.f};
        v8f c1 = c0;
        #pragma unroll
        for (int c = 0; c < 2; ++c) {
            v16bf bh = *(const v16bf*)(s_eh + code * DIM + c * 32 + koff);
            v16bf bl = *(const v16bf*)(s_el + code * DIM + c * 32 + koff);
            c0 = __builtin_amdgcn_wmma_f32_16x16x32_bf16(false, ah[0][c], false, bh, (short)0, c0, false, false);
            c1 = __builtin_amdgcn_wmma_f32_16x16x32_bf16(false, ah[1][c], false, bh, (short)0, c1, false, false);
            c0 = __builtin_amdgcn_wmma_f32_16x16x32_bf16(false, al[0][c], false, bh, (short)0, c0, false, false);
            c1 = __builtin_amdgcn_wmma_f32_16x16x32_bf16(false, al[1][c], false, bh, (short)0, c1, false, false);
            c0 = __builtin_amdgcn_wmma_f32_16x16x32_bf16(false, ah[0][c], false, bl, (short)0, c0, false, false);
            c1 = __builtin_amdgcn_wmma_f32_16x16x32_bf16(false, ah[1][c], false, bl, (short)0, c1, false, false);
        }
        float en   = s_enorm[code];
        float cidx = (float)code;
        #pragma unroll
        for (int v = 0; v < 8; ++v) {
            float s0 = en - 2.0f * c0[v];
            if (s0 < best0[v]) { best0[v] = s0; bidx0[v] = cidx; }
            float s1 = en - 2.0f * c1[v];
            if (s1 < best1[v]) { best1[v] = s1; bidx1[v] = cidx; }
        }
    }

    float ss = 0.0f;
    epilogue_tile(best0, bidx0, zn0, wave_tok0,      half, l16, z, emb, out, hist, acc, ss);
    epilogue_tile(best1, bidx1, zn1, wave_tok0 + 16, half, l16, z, emb, out, hist, acc, ss);
    #pragma unroll
    for (int off = 16; off >= 1; off >>= 1) ss += __shfl_xor(ss, off, 32);
    if (lane == 0) atomicAdd(&acc[0], ss);
}

__global__ void vq_finalize(const float* __restrict__ ws, float* __restrict__ out) {
    __shared__ float red[256];
    int tid = threadIdx.x;
    float ent = 0.0f;
    for (int i = tid; i < NUM_EMB; i += 256) {
        float p = ws[i] * (1.0f / (float)N_TOKENS);
        ent += p * logf(p + 1e-10f);
    }
    red[tid] = ent;
    __syncthreads();
    for (int s = 128; s > 0; s >>= 1) {
        if (tid < s) red[tid] += red[tid + s];
        __syncthreads();
    }
    if (tid == 0) {
        out[OFF_PERP] = expf(-red[0]);
        out[OFF_LOSS] = (1.0f + COMMIT) * ws[NUM_EMB] *
                        (1.0f / ((float)N_TOKENS * (float)DIM));
        out[OFF_MIND] = ws[NUM_EMB + 1] * (1.0f / (float)N_TOKENS);
    }
}

extern "C" void kernel_launch(void* const* d_in, const int* in_sizes, int n_in,
                              void* d_out, int out_size, void* d_ws, size_t ws_size,
                              hipStream_t stream) {
    const float* z   = (const float*)d_in[0];
    const float* emb = (const float*)d_in[1];
    float* out = (float*)d_out;
    float* ws  = (float*)d_ws;

    const size_t lds_bytes = (size_t)NUM_EMB * DIM * 2 * sizeof(__bf16)  // hi+lo bf16
                           + (size_t)NUM_EMB * sizeof(float);            // norms  (~130 KB)
    hipFuncSetAttribute((const void*)vq_main,
                        hipFuncAttributeMaxDynamicSharedMemorySize, (int)lds_bytes);

    vq_zero_ws<<<3, 256, 0, stream>>>(ws);
    vq_main<<<N_TOKENS / TOK_PER_WG, 256, lds_bytes, stream>>>(z, emb, out, ws);
    vq_finalize<<<1, 256, 0, stream>>>(ws, out);
}
